// aten_native_multi_head_attention_67164698575247
// MI455X (gfx1250) — compile-verified, matrix operand order checked by reading, not run on hardware
//
#include <hip/hip_runtime.h>

// Problem constants (match the reference).
#define BB 2
#define SS 2048
#define EE 1024
#define HH 16
#define DD 64

typedef __bf16 bf16;
typedef __attribute__((ext_vector_type(16))) __bf16 v16bf;
typedef __attribute__((ext_vector_type(8)))  __bf16 v8bf;
typedef __attribute__((ext_vector_type(4)))  __bf16 v4bf;
typedef __attribute__((ext_vector_type(8)))  float  v8f;
typedef __attribute__((ext_vector_type(4)))  unsigned int u32x4;
typedef __attribute__((ext_vector_type(8)))  int i32x8;
typedef __attribute__((ext_vector_type(4)))  int i32x4;

// ---------------------------------------------------------------------------
// D = A(16x32 bf16) x B(32x16 bf16) + C(16x16 f32), wave32.
// ---------------------------------------------------------------------------
__device__ __forceinline__ v8f wmma_bf16(v16bf a, v16bf b, v8f c) {
  return __builtin_amdgcn_wmma_f32_16x16x32_bf16(
      /*neg_a=*/false, a, /*neg_b=*/false, b,
      /*c_mod=*/(short)0, c, /*reuse_a=*/false, /*reuse_b=*/false);
}

// Load a 16x32 bf16 fragment from a row-major source (leading dim `ld`,
// K contiguous).  ISA layout (CDNA5 ISA, 16-bit A 16x32): lanes 0-15 hold
// K = k0+[0..7] and k0+[16..23]; lanes 16-31 hold K = k0+[8..15]/[24..31].
__device__ __forceinline__ v16bf load_frag(const bf16* p, int ld, int rc0, int k0) {
  const int lane = threadIdx.x & 31;
  const int idx  = lane & 15;
  const int hv   = lane >> 4;
  const bf16* row = p + (long)(rc0 + idx) * ld + k0 + (hv ? 8 : 0);
  v8bf lo = *(const v8bf*)(row);        // global_load_b128 / ds_load_b128
  v8bf hi = *(const v8bf*)(row + 16);
  return __builtin_shufflevector(lo, hi, 0,1,2,3,4,5,6,7,8,9,10,11,12,13,14,15);
}

// ---------------------------------------------------------------------------
// Tensor Data Mover: DMA a 2D tile (tile_d0 contiguous bf16 x tile_d1 rows,
// row stride = stride_elems) from global into LDS at lds_off.
// D# per CDNA5 ISA sect.8 (2-D tensor: groups 2/3 zero).
// This toolchain exposes the 6-arg builtin:
//   (u32x4 g0, i32x8 g1, i32x4 g2, i32x4 g3, i32x8 g4, i32 cpol)
// ---------------------------------------------------------------------------
__device__ __forceinline__ void tdm_load_tile_2d(unsigned lds_off, const void* gsrc,
                                                 unsigned tile_d0, unsigned tile_d1,
                                                 unsigned stride_elems) {
  const unsigned long long ga = (unsigned long long)gsrc;
  u32x4 g0;
  g0[0] = 1u;                                            // count=1 (valid D#)
  g0[1] = lds_off;                                       // lds_addr (bytes)
  g0[2] = (unsigned)ga;                                  // global_addr[31:0]
  g0[3] = ((unsigned)(ga >> 32) & 0x01FFFFFFu)           // global_addr[56:32]
          | (2u << 30);                                  // type=2 ("image")
  i32x8 g1;
  g1[0] = 0x00010000;                                    // data_size=1 (2 bytes)
  g1[1] = (int)((stride_elems & 0xFFFFu) << 16);         // tensor_dim0 lo16
  g1[2] = (int)((stride_elems >> 16) & 0xFFFFu);         // tensor_dim0 hi16, dim1 lo16=0
  g1[3] = (int)(16u | (tile_d0 << 16));                  // tensor_dim1 = 1<<20, tile_dim0
  g1[4] = (int)tile_d1;                                  // tile_dim1 (tile_dim2 = 0)
  g1[5] = (int)stride_elems;                             // tensor_dim0_stride lo32
  g1[6] = 0;                                             // stride hi, dim1_stride lo
  g1[7] = 0;
  i32x4 z4;
  z4[0] = 0; z4[1] = 0; z4[2] = 0; z4[3] = 0;
  i32x8 z8;
  z8[0] = 0; z8[1] = 0; z8[2] = 0; z8[3] = 0;
  z8[4] = 0; z8[5] = 0; z8[6] = 0; z8[7] = 0;
  __builtin_amdgcn_tensor_load_to_lds(g0, g1, z4, z4, z8, 0);
}

// ---------------------------------------------------------------------------
// Kernel 0: f32 -> bf16 conversion (bandwidth-bound pre-pass).
// ---------------------------------------------------------------------------
__global__ void k_cvt_bf16(const float* __restrict__ in, bf16* __restrict__ out, long n) {
  long i = ((long)blockIdx.x * blockDim.x + threadIdx.x) * 4;
  if (i + 3 < n) {
    const float4 f = *(const float4*)(in + i);   // global_load_b128
    v4bf o;
    o[0] = (bf16)f.x; o[1] = (bf16)f.y; o[2] = (bf16)f.z; o[3] = (bf16)f.w;
    *(v4bf*)(out + i) = o;                       // global_store_b64
  }
}

// ---------------------------------------------------------------------------
// Shared GEMM core: y[128x64] tile = x[128xE] @ w[64xE]^T (K-contiguous both).
// 8 waves; W tile TDM-staged into double-buffered LDS, shared by all waves.
// acc[j] = 16x16 f32 C fragment for columns n0+16j.
// ---------------------------------------------------------------------------
__device__ __forceinline__ void gemm_tile(const bf16* __restrict__ x,
                                          const bf16* __restrict__ w,
                                          bf16* lds_w /* [2][64*32] */,
                                          int m0, int n0, v8f* acc) {
  const int wid = threadIdx.x >> 5;

#pragma unroll
  for (int j = 0; j < 4; ++j)
#pragma unroll
    for (int r = 0; r < 8; ++r) acc[j][r] = 0.f;

  // Prologue: DMA first W tile (64 rows x 32 K) into buffer 0.
  if (wid == 0) {
    tdm_load_tile_2d((unsigned)(unsigned long long)(void*)lds_w,
                     w + (long)n0 * EE, 32, 64, EE);
    __builtin_amdgcn_s_wait_tensorcnt(0);
  }
  __syncthreads();

  for (int k0 = 0; k0 < EE; k0 += 32) {
    const int cur = (k0 >> 5) & 1;
    bf16* bufc = lds_w + cur * (64 * 32);
    bf16* bufn = lds_w + (cur ^ 1) * (64 * 32);

    // Kick off DMA of the next W tile while computing on the current one.
    const bool more = (k0 + 32) < EE;
    if (wid == 0 && more) {
      tdm_load_tile_2d((unsigned)(unsigned long long)(void*)bufn,
                       w + (long)n0 * EE + k0 + 32, 32, 64, EE);
    }

    v16bf a = load_frag(x, EE, m0, k0);          // per-wave rows, global
#pragma unroll
    for (int j = 0; j < 4; ++j) {
      v16bf bfr = load_frag(bufc, 32, j * 16, 0);  // shared W tile, LDS
      acc[j] = wmma_bf16(a, bfr, acc[j]);
    }

    if (wid == 0 && more) __builtin_amdgcn_s_wait_tensorcnt(0);
    __syncthreads();
  }
}

// ---------------------------------------------------------------------------
// Kernel 1: fused QKV projection.  y = x @ W_z^T + b_z, stored bf16 in
// [B,H,S,D] layout.  Grid: (M/128, E/64, 3), block 256 (8 waves).
// ---------------------------------------------------------------------------
__global__ void k_gemm_qkv(const bf16* __restrict__ xq, const bf16* __restrict__ xk,
                           const bf16* __restrict__ xv, const bf16* __restrict__ w,
                           const float* __restrict__ bias,
                           bf16* __restrict__ Qb, bf16* __restrict__ Kb,
                           bf16* __restrict__ Vb) {
  __shared__ bf16 lds_w[2 * 64 * 32];

  const int z = blockIdx.z;
  const bf16* x  = (z == 0) ? xq : (z == 1) ? xk : xv;
  const bf16* wz = w + (long)z * EE * EE;
  const float* bz = bias + z * EE;
  bf16* dst = (z == 0) ? Qb : (z == 1) ? Kb : Vb;

  const int wid  = threadIdx.x >> 5;
  const int lane = threadIdx.x & 31;
  const int idx  = lane & 15;
  const int hv   = lane >> 4;
  const int m0 = blockIdx.x * 128 + wid * 16;
  const int n0 = blockIdx.y * 64;

  v8f acc[4];
  gemm_tile(x, wz, lds_w, m0, n0, acc);

#pragma unroll
  for (int j = 0; j < 4; ++j) {
    const int f = n0 + j * 16 + idx;
    const float bval = bz[f];
    const int h = f >> 6;      // f / D
    const int d = f & 63;      // f % D
#pragma unroll
    for (int r = 0; r < 8; ++r) {
      const int m = m0 + r + 8 * hv;   // C layout: VGPR r = row r (+8 upper half)
      const int b = m >> 11;           // m / S
      const int s = m & (SS - 1);      // m % S
      dst[(((long)(b * HH + h)) * SS + s) * DD + d] = (bf16)(acc[j][r] + bval);
    }
  }
}

// ---------------------------------------------------------------------------
// Kernel 2: flash attention.  Grid: (S/64, B*H), block 128 (4 waves).
// Each wave owns 16 query rows; key loop in tiles of 32 with online softmax.
// ---------------------------------------------------------------------------
__global__ void k_attn(const bf16* __restrict__ Qb, const bf16* __restrict__ Kb,
                       const bf16* __restrict__ Vb, const float* __restrict__ mask,
                       bf16* __restrict__ ctx) {
  __shared__ bf16 lds_vt[DD * 32];       // V tile transposed: [d][t], 4 KB
  __shared__ bf16 lds_p[4 * 16 * 32];    // per-wave P tile, row-major 16x32

  const int wid  = threadIdx.x >> 5;
  const int lane = threadIdx.x & 31;
  const int idx  = lane & 15;
  const int hv   = lane >> 4;
  const int bh   = blockIdx.y;
  const long hbase = (long)bh * SS * DD;
  const int qm0 = blockIdx.x * 64 + wid * 16;
  const float scale = 0.125f;            // 1/sqrt(64)

  // Q fragments for this wave (D = 64 = 2 K-steps), loaded once.
  const v16bf aq0 = load_frag(Qb + hbase, DD, qm0, 0);
  const v16bf aq1 = load_frag(Qb + hbase, DD, qm0, 32);

  float mrun[8], lrun[8];
  v8f acc[4];
#pragma unroll
  for (int r = 0; r < 8; ++r) { mrun[r] = -3.0e38f; lrun[r] = 0.f; }
#pragma unroll
  for (int j = 0; j < 4; ++j)
#pragma unroll
    for (int r = 0; r < 8; ++r) acc[j][r] = 0.f;

  for (int t0 = 0; t0 < SS; t0 += 32) {
    // Speculative prefetch of next iteration's K/V tiles (global_prefetch_b8).
    if (t0 + 32 < SS) {
      __builtin_prefetch(Kb + hbase + (long)(t0 + 32) * DD, 0, 0);
      __builtin_prefetch(Vb + hbase + (long)(t0 + 32) * DD, 0, 0);
    }

    // --- cooperatively stage V tile (32 keys x 64 dims) transposed into LDS
    __syncthreads();
    {
      const int tl = threadIdx.x >> 2;          // key row 0..31
      const int db = (threadIdx.x & 3) << 4;    // dim base 0/16/32/48
      const bf16* vs = Vb + hbase + (long)(t0 + tl) * DD + db;
      v8bf v0 = *(const v8bf*)(vs);
      v8bf v1 = *(const v8bf*)(vs + 8);
#pragma unroll
      for (int j = 0; j < 8; ++j) {
        lds_vt[(db + j) * 32 + tl]     = v0[j];
        lds_vt[(db + 8 + j) * 32 + tl] = v1[j];
      }
    }
    __syncthreads();

    // --- scores S = (Q K^T) * scale + mask, two 16-key sub-tiles
    float s0[8], s1[8];
#pragma unroll
    for (int jt = 0; jt < 2; ++jt) {
      v16bf bk0 = load_frag(Kb + hbase, DD, t0 + jt * 16, 0);
      v16bf bk1 = load_frag(Kb + hbase, DD, t0 + jt * 16, 32);
      v8f sc;
#pragma unroll
      for (int r = 0; r < 8; ++r) sc[r] = 0.f;
      sc = wmma_bf16(aq0, bk0, sc);
      sc = wmma_bf16(aq1, bk1, sc);
      const int col = t0 + jt * 16 + idx;
      float* sout = jt ? s1 : s0;
#pragma unroll
      for (int r = 0; r < 8; ++r) {
        const int qrow = qm0 + r + 8 * hv;
        sout[r] = sc[r] * scale + mask[(long)qrow * SS + col];
      }
    }

    // --- online softmax per row (rows live in 16-lane halves; xor masks
    //     1/2/4/8 reduce across the row's 16 columns without crossing halves)
    float corr[8];
#pragma unroll
    for (int r = 0; r < 8; ++r) {
      float tm = fmaxf(s0[r], s1[r]);
#pragma unroll
      for (int m = 1; m < 16; m <<= 1) tm = fmaxf(tm, __shfl_xor(tm, m, 32));
      const float mn = fmaxf(mrun[r], tm);
      corr[r] = __expf(mrun[r] - mn);
      const float p0 = __expf(s0[r] - mn);
      const float p1 = __expf(s1[r] - mn);
      float rs = p0 + p1;
#pragma unroll
      for (int m = 1; m < 16; m <<= 1) rs += __shfl_xor(rs, m, 32);
      lrun[r] = lrun[r] * corr[r] + rs;
      mrun[r] = mn;
      s0[r] = p0;
      s1[r] = p1;
    }
#pragma unroll
    for (int j = 0; j < 4; ++j)
#pragma unroll
      for (int r = 0; r < 8; ++r) acc[j][r] *= corr[r];

    // --- re-layout P (C fragment -> A fragment) through per-wave LDS
    bf16* pl = lds_p + wid * (16 * 32);
#pragma unroll
    for (int r = 0; r < 8; ++r) {
      pl[(r + 8 * hv) * 32 + idx]      = (bf16)s0[r];
      pl[(r + 8 * hv) * 32 + 16 + idx] = (bf16)s1[r];
    }
    const v16bf ap = load_frag(pl, 32, 0, 0);

    // --- ctx += P (16x32) x V (32x64), 4 sub-tiles over D
#pragma unroll
    for (int j = 0; j < 4; ++j) {
      v16bf bv = load_frag(lds_vt, 32, j * 16, 0);
      acc[j] = wmma_bf16(ap, bv, acc[j]);
    }
  }

  // --- normalize and emit ctx in [B,S,E] (bf16) for the output projection
  const int b = bh / HH;
  const int h = bh % HH;
#pragma unroll
  for (int r = 0; r < 8; ++r) {
    const float inv = 1.f / lrun[r];
    const int srow = qm0 + r + 8 * hv;
#pragma unroll
    for (int j = 0; j < 4; ++j) {
      ctx[((long)(b * SS + srow)) * EE + h * DD + j * 16 + idx] =
          (bf16)(acc[j][r] * inv);
    }
  }
}

// ---------------------------------------------------------------------------
// Kernel 3: output projection.  out = ctx @ Wp^T + bp (f32 result).
// Grid: (M/128, E/64), block 256.
// ---------------------------------------------------------------------------
__global__ void k_gemm_out(const bf16* __restrict__ x, const bf16* __restrict__ w,
                           const float* __restrict__ bias, float* __restrict__ out) {
  __shared__ bf16 lds_w[2 * 64 * 32];

  const int wid  = threadIdx.x >> 5;
  const int lane = threadIdx.x & 31;
  const int idx  = lane & 15;
  const int hv   = lane >> 4;
  const int m0 = blockIdx.x * 128 + wid * 16;
  const int n0 = blockIdx.y * 64;

  v8f acc[4];
  gemm_tile(x, w, lds_w, m0, n0, acc);

#pragma unroll
  for (int j = 0; j < 4; ++j) {
    const int f = n0 + j * 16 + idx;
    const float bval = bias[f];
#pragma unroll
    for (int r = 0; r < 8; ++r) {
      const int m = m0 + r + 8 * hv;
      out[(long)m * EE + f] = acc[j][r] + bval;
    }
  }
}

// ---------------------------------------------------------------------------
// Host launcher.
// ---------------------------------------------------------------------------
extern "C" void kernel_launch(void* const* d_in, const int* in_sizes, int n_in,
                              void* d_out, int out_size, void* d_ws, size_t ws_size,
                              hipStream_t stream) {
  const float* query = (const float*)d_in[0];
  const float* key   = (const float*)d_in[1];
  const float* value = (const float*)d_in[2];
  const float* qkvw  = (const float*)d_in[3];
  const float* qkvb  = (const float*)d_in[4];
  const float* projw = (const float*)d_in[5];
  const float* projb = (const float*)d_in[6];
  const float* mask  = (const float*)d_in[7];
  float* out = (float*)d_out;

  const long NX    = (long)BB * SS * EE;      // 4,194,304
  const long NWQKV = 3L * EE * EE;            // 3,145,728
  const long NWP   = (long)EE * EE;           // 1,048,576

  // Workspace layout (bf16 elements, ~67 MB total)
  bf16* ws    = (bf16*)d_ws;
  bf16* xq    = ws;
  bf16* xk    = xq + NX;
  bf16* xv    = xk + NX;
  bf16* wqkv  = xv + NX;
  bf16* wproj = wqkv + NWQKV;
  bf16* Qb    = wproj + NWP;
  bf16* Kb    = Qb + NX;
  bf16* Vb    = Kb + NX;
  bf16* ctx   = Vb + NX;

  // 0) f32 -> bf16 conversions
  const int CT = 256;
  k_cvt_bf16<<<(int)(NX / 4 / CT), CT, 0, stream>>>(query, xq, NX);
  k_cvt_bf16<<<(int)(NX / 4 / CT), CT, 0, stream>>>(key, xk, NX);
  k_cvt_bf16<<<(int)(NX / 4 / CT), CT, 0, stream>>>(value, xv, NX);
  k_cvt_bf16<<<(int)(NWQKV / 4 / CT), CT, 0, stream>>>(qkvw, wqkv, NWQKV);
  k_cvt_bf16<<<(int)(NWP / 4 / CT), CT, 0, stream>>>(projw, wproj, NWP);

  // 1) fused QKV projection -> bf16 [B,H,S,D]
  k_gemm_qkv<<<dim3((BB * SS) / 128, EE / 64, 3), 256, 0, stream>>>(
      xq, xk, xv, wqkv, qkvb, Qb, Kb, Vb);

  // 2) flash attention -> bf16 ctx [B,S,E]
  k_attn<<<dim3(SS / 64, BB * HH), 128, 0, stream>>>(Qb, Kb, Vb, mask, ctx);

  // 3) output projection -> f32 [B,S,E]
  k_gemm_out<<<dim3((BB * SS) / 128, EE / 64), 256, 0, stream>>>(
      ctx, wproj, projb, out);
}
